// GraphSAGE_LinkPredictor_18348100288600
// MI455X (gfx1250) — matrix operand order repaired and checked by reading, not compiled, not run on hardware
//
#include <hip/hip_runtime.h>
#include <hip/hip_bf16.h>
#include <stdint.h>

typedef __attribute__((ext_vector_type(16))) _Float16 v16h;
typedef __attribute__((ext_vector_type(8)))  float    v8f;

#define NCH 128

// ---------------- utility kernels ----------------

__global__ void zero_kernel(float* __restrict__ p, long long n) {
    long long i = (long long)blockIdx.x * blockDim.x + threadIdx.x;
    if (i < n) p[i] = 0.0f;
}

__global__ void deg_kernel(const int* __restrict__ dst, float* __restrict__ deg, int E) {
    int e = blockIdx.x * blockDim.x + threadIdx.x;
    if (e < E) atomicAdd(deg + dst[e], 1.0f);
}

__global__ void rdeg_kernel(const float* __restrict__ deg, float* __restrict__ rdeg, int n) {
    int i = blockIdx.x * blockDim.x + threadIdx.x;
    if (i < n) rdeg[i] = 1.0f / fmaxf(deg[i], 1.0f);
}

// agg[dst] += x[src]  (one thread per (edge, 4-channel group)); T = float or _Float16
template <typename T>
__global__ void scatter_kernel(const T* __restrict__ x, const int* __restrict__ src,
                               const int* __restrict__ dst, float* __restrict__ agg, int E) {
    long long gid = (long long)blockIdx.x * blockDim.x + threadIdx.x;
    if (gid >= (long long)E * (NCH / 4)) return;
    int e  = (int)(gid >> 5);          // NCH/4 == 32 groups per edge
    int c4 = (int)(gid & 31) * 4;
    const T* sp = x + (long long)src[e] * NCH + c4;
    float* dptr = agg + (long long)dst[e] * NCH + c4;
    atomicAdd(dptr + 0, (float)sp[0]);
    atomicAdd(dptr + 1, (float)sp[1]);
    atomicAdd(dptr + 2, (float)sp[2]);
    atomicAdd(dptr + 3, (float)sp[3]);
}

// ---------------- weight pre-pack: f32 [K x 128] -> f16 WMMA B fragments ----------------
// packed[((kt*8 + ct)*32 + lane)*16 + i] = W[(kt*32 + ((i&8)<<1) + (lane>>4)*8 + (i&7))*128 + ct*16 + (lane&15)]
// Each lane's 16 values land in 32 contiguous bytes -> 2x global_load_b128 in the GEMM.

__global__ void pack_w_kernel(const float* __restrict__ W, _Float16* __restrict__ out, int K) {
    int idx = blockIdx.x * blockDim.x + threadIdx.x;
    if (idx >= K * NCH) return;
    int i    = idx & 15;
    int lane = (idx >> 4) & 31;
    int ct   = (idx >> 9) & 7;
    int kt   = idx >> 12;
    int g = lane >> 4, l = lane & 15;
    int kb = kt * 32 + ((i & 8) << 1) + g * 8 + (i & 7);
    out[idx] = (_Float16)W[kb * NCH + ct * 16 + l];
}

// ---------------- SAGE conv GEMM: out(f16) = relu( (agg*rdeg)@Wl + x@Wr + b ) -----------
// grid.x = N/16 row tiles, 256 threads = 8 waves, wave w owns columns [16w,16w+16).
// TIN = float (conv1 root input x) or _Float16 (conv2 root input h).

template <typename TIN>
__global__ __launch_bounds__(256) void sage_gemm_kernel(
    const float* __restrict__ agg, const float* __restrict__ rdeg,
    const TIN* __restrict__ xin,
    const v16h* __restrict__ Wlp, const v16h* __restrict__ Wrp,
    const float* __restrict__ bias, _Float16* __restrict__ out, int Nn)
{
    const int lane = threadIdx.x & 31;
    const int wave = threadIdx.x >> 5;
    const int g    = lane >> 4;
    const int l    = lane & 15;
    const int m0   = blockIdx.x * 16;
    const int col0 = wave * 16;
    int row = m0 + l; if (row >= Nn) row = Nn - 1;   // clamp, keep EXEC all-ones
    const float rd = rdeg[row];

    const float* arow0 = agg + (long long)row * NCH;
    const TIN*   arow1 = xin + (long long)row * NCH;

    v8f acc = {};
    // pass 0: mean @ Wl  (f32 aggregate, scaled by 1/deg, converted to f16)
#pragma unroll
    for (int kk = 0; kk < 4; ++kk) {
        v16h a;
#pragma unroll
        for (int i = 0; i < 16; ++i) {
            int ka = kk * 32 + ((i & 8) << 1) + g * 8 + (i & 7);
            a[i] = (_Float16)(arow0[ka] * rd);
        }
        v16h b = Wlp[(kk * 8 + wave) * 32 + lane];
        acc = __builtin_amdgcn_wmma_f32_16x16x32_f16(false, a, false, b,
                                                     (short)0, acc, false, false);
    }
    // pass 1: x @ Wr
#pragma unroll
    for (int kk = 0; kk < 4; ++kk) {
        v16h a;
#pragma unroll
        for (int i = 0; i < 16; ++i) {
            int ka = kk * 32 + ((i & 8) << 1) + g * 8 + (i & 7);
            a[i] = (_Float16)arow1[ka];
        }
        v16h b = Wrp[(kk * 8 + wave) * 32 + lane];
        acc = __builtin_amdgcn_wmma_f32_16x16x32_f16(false, a, false, b,
                                                     (short)0, acc, false, false);
    }

    const float bv = bias[col0 + l];
#pragma unroll
    for (int r = 0; r < 8; ++r) {
        int m = m0 + r + 8 * g;
        if (m < Nn) {
            float v = acc[r] + bv;
            out[(long long)m * NCH + col0 + l] = (_Float16)(v > 0.0f ? v : 0.0f);
        }
    }
}

// ---------------- decode: out[p] = relu(concat(z[u],z[v],ea) @ Wm1 + bm1) @ Wm2 + bm2 ----

__global__ __launch_bounds__(256) void decode_kernel(
    const _Float16* __restrict__ z, const int* __restrict__ ep, int P,
    const float* __restrict__ ea,
    const v16h* __restrict__ Wm1p, const float* __restrict__ bm1,
    const float* __restrict__ Wm2, const float* __restrict__ bm2,
    float* __restrict__ out)
{
    __shared__ float red[16];
    const int lane = threadIdx.x & 31;
    const int wave = threadIdx.x >> 5;
    const int g    = lane >> 4;
    const int l    = lane & 15;
    const int p0   = blockIdx.x * 16;
    const int col0 = wave * 16;

    if (threadIdx.x < 16) red[threadIdx.x] = 0.0f;
    __syncthreads();

    int p = p0 + l; if (p >= P) p = P - 1;
    const long long uu = ep[p];
    const long long vv = ep[(long long)P + p];
    const _Float16* zu = z  + uu * NCH;
    const _Float16* zv = z  + vv * NCH;
    const float*    er = ea + (long long)p * NCH;

    v8f acc = {};
    // K segments 0..255: gathered f16 endpoint embeddings (no conversion needed)
#pragma unroll
    for (int kk = 0; kk < 8; ++kk) {
        const _Float16* arow = (kk < 4) ? (zu + kk * 32) : (zv + (kk - 4) * 32);
        v16h a;
#pragma unroll
        for (int i = 0; i < 16; ++i) {
            int ka = ((i & 8) << 1) + g * 8 + (i & 7);
            a[i] = arow[ka];
        }
        v16h b = Wm1p[(kk * 8 + wave) * 32 + lane];
        acc = __builtin_amdgcn_wmma_f32_16x16x32_f16(false, a, false, b,
                                                     (short)0, acc, false, false);
    }
    // K segments 256..383: f32 edge attributes
#pragma unroll
    for (int kk = 8; kk < 12; ++kk) {
        const float* arow = er + (kk - 8) * 32;
        v16h a;
#pragma unroll
        for (int i = 0; i < 16; ++i) {
            int ka = ((i & 8) << 1) + g * 8 + (i & 7);
            a[i] = (_Float16)arow[ka];
        }
        v16h b = Wm1p[(kk * 8 + wave) * 32 + lane];
        acc = __builtin_amdgcn_wmma_f32_16x16x32_f16(false, a, false, b,
                                                     (short)0, acc, false, false);
    }

    const float b1 = bm1[col0 + l];
    const float w2 = Wm2[col0 + l];
#pragma unroll
    for (int r = 0; r < 8; ++r) {
        float h = acc[r] + b1;
        h = h > 0.0f ? h : 0.0f;
        atomicAdd(&red[r + 8 * g], h * w2);   // ds_add_f32 reduction across 8 waves
    }
    __syncthreads();
    if (threadIdx.x < 16) {
        int pw = p0 + threadIdx.x;
        if (pw < P) out[pw] = red[threadIdx.x] + bm2[0];
    }
}

// ---------------- host-side orchestration ----------------

static inline char* ws_align(char* p) {
    return (char*)(((uintptr_t)p + 255) & ~(uintptr_t)255);
}

extern "C" void kernel_launch(void* const* d_in, const int* in_sizes, int n_in,
                              void* d_out, int out_size, void* d_ws, size_t ws_size,
                              hipStream_t stream) {
    const float* x   = (const float*)d_in[0];
    const int*   ei  = (const int*)  d_in[1];
    const int*   ep  = (const int*)  d_in[2];
    const float* ea  = (const float*)d_in[3];
    const float* W1l = (const float*)d_in[4];
    const float* W1r = (const float*)d_in[5];
    const float* b1  = (const float*)d_in[6];
    const float* W2l = (const float*)d_in[7];
    const float* W2r = (const float*)d_in[8];
    const float* b2  = (const float*)d_in[9];
    const float* Wm1 = (const float*)d_in[10];
    const float* bm1 = (const float*)d_in[11];
    const float* Wm2 = (const float*)d_in[12];
    const float* bm2 = (const float*)d_in[13];
    float* out = (float*)d_out;

    const int N = in_sizes[0] / NCH;
    const int E = in_sizes[1] / 2;
    const int P = in_sizes[2] / 2;
    const int* src = ei;
    const int* dst = ei + E;

    char* w = (char*)d_ws;
    float*    agg  = (float*)w;    w = ws_align(w + (size_t)N * NCH * sizeof(float));
    float*    deg  = (float*)w;    w = ws_align(w + (size_t)N * sizeof(float));
    float*    rdeg = (float*)w;    w = ws_align(w + (size_t)N * sizeof(float));
    _Float16* h    = (_Float16*)w; w = ws_align(w + (size_t)N * NCH * sizeof(_Float16));
    _Float16* z    = (_Float16*)w; w = ws_align(w + (size_t)N * NCH * sizeof(_Float16));
    _Float16* pW1l = (_Float16*)w; w = ws_align(w + (size_t)NCH * NCH * sizeof(_Float16));
    _Float16* pW1r = (_Float16*)w; w = ws_align(w + (size_t)NCH * NCH * sizeof(_Float16));
    _Float16* pW2l = (_Float16*)w; w = ws_align(w + (size_t)NCH * NCH * sizeof(_Float16));
    _Float16* pW2r = (_Float16*)w; w = ws_align(w + (size_t)NCH * NCH * sizeof(_Float16));
    _Float16* pWm1 = (_Float16*)w; w = ws_align(w + (size_t)3 * NCH * NCH * sizeof(_Float16));

    const long long featN = (long long)N * NCH;
    const long long scatW = (long long)E * (NCH / 4);
    dim3 blk(256);
    dim3 gZeroF((unsigned)((featN + 255) / 256));
    dim3 gZeroN((unsigned)((N + 255) / 256));
    dim3 gScat((unsigned)((scatW + 255) / 256));
    dim3 gEdge((unsigned)((E + 255) / 256));
    dim3 gGemm((unsigned)((N + 15) / 16));
    dim3 gDec((unsigned)((P + 15) / 16));
    dim3 gPackS((unsigned)((NCH * NCH + 255) / 256));      // 128x128 weights
    dim3 gPackM((unsigned)((3 * NCH * NCH + 255) / 256));  // 384x128 weight

    // pre-pack all weight matrices into WMMA B-fragment layout (f16)
    pack_w_kernel<<<gPackS, blk, 0, stream>>>(W1l, pW1l, NCH);
    pack_w_kernel<<<gPackS, blk, 0, stream>>>(W1r, pW1r, NCH);
    pack_w_kernel<<<gPackS, blk, 0, stream>>>(W2l, pW2l, NCH);
    pack_w_kernel<<<gPackS, blk, 0, stream>>>(W2r, pW2r, NCH);
    pack_w_kernel<<<gPackM, blk, 0, stream>>>(Wm1, pWm1, 3 * NCH);

    // degree (depends only on dst — compute once, reuse for both convs)
    zero_kernel<<<gZeroN, blk, 0, stream>>>(deg, N);
    deg_kernel<<<gEdge, blk, 0, stream>>>(dst, deg, E);
    rdeg_kernel<<<gZeroN, blk, 0, stream>>>(deg, rdeg, N);

    // conv 1: h = relu(mean(x)@W1l + b1 + x@W1r)   (f32 inputs, f16 output)
    zero_kernel<<<gZeroF, blk, 0, stream>>>(agg, featN);
    scatter_kernel<float><<<gScat, blk, 0, stream>>>(x, src, dst, agg, E);
    sage_gemm_kernel<float><<<gGemm, blk, 0, stream>>>(agg, rdeg, x,
                                                       (const v16h*)pW1l, (const v16h*)pW1r,
                                                       b1, h, N);

    // conv 2: z = relu(mean(h)@W2l + b2 + h@W2r)   (f16 input h, f16 output)
    zero_kernel<<<gZeroF, blk, 0, stream>>>(agg, featN);
    scatter_kernel<_Float16><<<gScat, blk, 0, stream>>>(h, src, dst, agg, E);
    sage_gemm_kernel<_Float16><<<gGemm, blk, 0, stream>>>(agg, rdeg, h,
                                                          (const v16h*)pW2l, (const v16h*)pW2r,
                                                          b2, z, N);

    // decode
    decode_kernel<<<gDec, blk, 0, stream>>>(z, ep, P, ea,
                                            (const v16h*)pWm1, bm1, Wm2, bm2, out);
}
